// SubLSTM_27865747816989
// MI455X (gfx1250) — matrix operand order repaired and checked by reading, not compiled, not run on hardware
//
#include <hip/hip_runtime.h>
#include <cstdint>
#include <cstddef>

// Problem constants (fixed by reference)
#define TT 512
#define BB 64
#define II 1024
#define HH 1024
#define NG 3
#define GH (NG * HH)      // 3072 gate columns
#define K2 (2 * HH)       // 2048 fused K (W | R)
#define NBLK 32           // persistent grid size

typedef __attribute__((ext_vector_type(16))) __bf16 v16bf;
typedef __attribute__((ext_vector_type(8)))  __bf16 v8bf;
typedef __attribute__((ext_vector_type(8)))  float  v8f;

__device__ __forceinline__ float sigm(float x) {
  return 1.0f / (1.0f + __expf(-x));
}

// A-fragment (16x32 bf16): lane L holds row (L&15);
// halves 0..15 = K (L>>4)*16 .. +15 (contiguous, matches B)
__device__ __forceinline__ v16bf load_a_frag(const __bf16* p) {
  v8bf lo = *(const v8bf*)(p);
  v8bf hi = *(const v8bf*)(p + 8);
  v16bf a;
#pragma unroll
  for (int i = 0; i < 8; ++i) { a[i] = lo[i]; a[i + 8] = hi[i]; }
  return a;
}

// Device-wide barrier for a persistent grid of NBLK blocks.
// Monotonic counter: barrier k complete when cnt >= (k+1)*NBLK.
__device__ __forceinline__ void grid_barrier(unsigned* cnt, unsigned& barno) {
  __syncthreads();
  if (threadIdx.x == 0) {
    __threadfence();                       // release prior writes to device scope
    atomicAdd(cnt, 1u);
    unsigned target = (barno + 1u) * NBLK;
    while (__hip_atomic_load(cnt, __ATOMIC_RELAXED, __HIP_MEMORY_SCOPE_AGENT) < target)
      __builtin_amdgcn_s_sleep(1);
    __threadfence();                       // acquire
  }
  __syncthreads();
  ++barno;
}

// ============================================================================
// One layer-step phase.
// Block = 512 threads = 16 waves: pos = wid&7 (mtile=pos&3 over 64 batch rows,
// ntile=pos>>2 over 32 hidden cols), kslice = wid>>3 splits K=2048 in half
// (kslice0: input half + bias, kslice1: recurrent half). Each wave runs a
// register double-buffered 32-iteration K loop with 3 WMMAs (i,o,z gates) on
// the SAME 16x16 (batch, unit) region; partials reduced through LDS; kslice0
// finishes with the fused subtractive-LSTM pointwise update in-register.
// ============================================================================
__device__ void layer_phase(
    const __bf16* __restrict__ a0,     // input half of A   [64 x 1024]
    const __bf16* __restrict__ a1,     // h_prev half of A  [64 x 1024]
    const __bf16* __restrict__ rw,     // fused weights     [3072 x 2048] bf16
    const float*  __restrict__ bias,   // bi + bh           [3072]
    const float*  __restrict__ fg,     // sigmoid(f)        [1024]
    float*        __restrict__ c,      // cell state        [64 x 1024]
    __bf16*       __restrict__ h_out,  // new h (bf16)      [64 x 1024]
    float*        __restrict__ out_f,  // fp32 h out (layer 1) or null
    float*        __restrict__ hfin,   // final-h dest or null
    float*        __restrict__ cfin,   // final-c dest or null
    float (*red)[32][32])              // LDS reduction buffer [8][32][32]
{
  const int lane  = threadIdx.x & 31;
  const int wid   = threadIdx.x >> 5;            // 0..15
  const int pos   = wid & 7;
  const int ks    = wid >> 3;                    // K-slice 0/1
  const int mtile = pos & 3;
  const int ntile = pos >> 2;
  const int u0    = blockIdx.x * 32 + ntile * 16;
  const int m0    = mtile * 16;
  const int lcol  = lane & 15;
  const int lhi   = lane >> 4;
  const int col   = u0 + lcol;

  // Accumulators: kslice0 carries bias, kslice1 starts at zero.
  v8f acc0, acc1, acc2;
  {
    float b0 = (ks == 0) ? bias[0 * HH + col] : 0.0f;
    float b1 = (ks == 0) ? bias[1 * HH + col] : 0.0f;
    float b2 = (ks == 0) ? bias[2 * HH + col] : 0.0f;
#pragma unroll
    for (int i = 0; i < 8; ++i) { acc0[i] = b0; acc1[i] = b1; acc2[i] = b2; }
  }

  const __bf16* abase = (ks == 0) ? a0 : a1;
  const size_t  koff  = (ks == 0) ? 0 : HH;      // column offset into [W|R]
  const __bf16* ap  = abase + (size_t)(m0 + lcol) * HH + lhi * 16;
  const __bf16* bp0 = rw + (size_t)(0 * HH + col) * K2 + koff + lhi * 16;
  const __bf16* bp1 = rw + (size_t)(1 * HH + col) * K2 + koff + lhi * 16;
  const __bf16* bp2 = rw + (size_t)(2 * HH + col) * K2 + koff + lhi * 16;

  // Software-pipelined K loop: fragments for kk+1 are in flight while the
  // 3 WMMAs for kk execute, so waits retire at loadcnt > 0.
  v16bf a_c  = load_a_frag(ap);
  v16bf b0_c = *(const v16bf*)(bp0);
  v16bf b1_c = *(const v16bf*)(bp1);
  v16bf b2_c = *(const v16bf*)(bp2);
#pragma unroll 4
  for (int kk = 0; kk < 32; ++kk) {
    const int kn = (kk + 1 < 32) ? kk + 1 : 31;
    __builtin_prefetch(bp0 + kn * 32 + 256, 0, 1);
    v16bf a_n  = load_a_frag(ap + kn * 32);
    v16bf b0_n = *(const v16bf*)(bp0 + kn * 32);
    v16bf b1_n = *(const v16bf*)(bp1 + kn * 32);
    v16bf b2_n = *(const v16bf*)(bp2 + kn * 32);
    acc0 = __builtin_amdgcn_wmma_f32_16x16x32_bf16(false, a_c, false, b0_c, (short)0, acc0, false, false);
    acc1 = __builtin_amdgcn_wmma_f32_16x16x32_bf16(false, a_c, false, b1_c, (short)0, acc1, false, false);
    acc2 = __builtin_amdgcn_wmma_f32_16x16x32_bf16(false, a_c, false, b2_c, (short)0, acc2, false, false);
    a_c = a_n; b0_c = b0_n; b1_c = b1_n; b2_c = b2_n;
  }

  // Cross-kslice reduction through LDS.
  if (ks == 1) {
    *(v8f*)&red[pos][lane][0]  = acc0;
    *(v8f*)&red[pos][lane][8]  = acc1;
    *(v8f*)&red[pos][lane][16] = acc2;
  }
  __syncthreads();
  if (ks == 0) {
    acc0 += *(const v8f*)&red[pos][lane][0];
    acc1 += *(const v8f*)&red[pos][lane][8];
    acc2 += *(const v8f*)&red[pos][lane][16];

    const float fgc = fg[col];
#pragma unroll
    for (int r = 0; r < 8; ++r) {
      int row = m0 + lhi * 8 + r;                // C/D layout: M = (lane>>4)*8 + vgpr
      size_t idx = (size_t)row * HH + col;
      float gi = sigm(acc0[r]);
      float go = sigm(acc1[r]);
      float gz = sigm(acc2[r]);
      float cn = c[idx] * fgc + gz - gi;
      float hn = sigm(cn) - go;
      c[idx] = cn;
      h_out[idx] = (__bf16)hn;
      if (out_f) out_f[idx] = hn;
      if (hfin)  { hfin[idx] = hn; cfin[idx] = cn; }
    }
  }
}

// ============================================================================
// Persistent kernel: all 512 timesteps x 2 layers in one launch.
// Cross-block sync via device-scope atomic barrier; weights stay hot in L2.
// ============================================================================
__global__ __launch_bounds__(512) void sublstm_persistent(
    const __bf16* __restrict__ xb,
    const __bf16* __restrict__ rw0, const __bf16* __restrict__ rw1,
    const float* __restrict__ bias0, const float* __restrict__ bias1,
    const float* __restrict__ fg0, const float* __restrict__ fg1,
    float* __restrict__ c0, float* __restrict__ c1,
    __bf16* __restrict__ h0a, __bf16* __restrict__ h0b,
    __bf16* __restrict__ h1a, __bf16* __restrict__ h1b,
    float* __restrict__ out, unsigned* __restrict__ cnt)
{
  __shared__ __align__(32) float red[8][32][32];   // 32 KB reduction buffer
  unsigned barno = 0;
  float* hfin = out + (size_t)TT * BB * HH;        // h_final [L,B,H]
  float* cfin = hfin + (size_t)2 * BB * HH;        // c_final [L,B,H]

  for (int t = 0; t < TT; ++t) {
    const int rd = t & 1;
    const __bf16* h0r = rd ? h0b : h0a;  __bf16* h0w = rd ? h0a : h0b;
    const __bf16* h1r = rd ? h1b : h1a;  __bf16* h1w = rd ? h1a : h1b;
    const bool last = (t == TT - 1);

    // Layer 0: A = [x_t | h0_prev]
    layer_phase(xb + (size_t)t * BB * II, h0r, rw0, bias0, fg0, c0, h0w,
                nullptr,
                last ? hfin : nullptr,
                last ? cfin : nullptr, red);
    grid_barrier(cnt, barno);              // publish h0_new

    // Layer 1: A = [h0_new | h1_prev]; writes fp32 output slice
    layer_phase(h0w, h1r, rw1, bias1, fg1, c1, h1w,
                out + (size_t)t * BB * HH,
                last ? hfin + (size_t)BB * HH : nullptr,
                last ? cfin + (size_t)BB * HH : nullptr, red);
    grid_barrier(cnt, barno);              // publish h1_new before next step
  }
}

// ============================================================================
// One-time prep kernels
// ============================================================================
__global__ void f32_to_bf16(const float* __restrict__ s, __bf16* __restrict__ d, int n) {
  int i = blockIdx.x * blockDim.x + threadIdx.x;
  if (i < n) d[i] = (__bf16)s[i];
}

// dst[n][0..K-1] = W[n][k] ; dst[n][K..2K-1] = R[n][k]   (bf16, row-major)
__global__ void pack_rw(const float* __restrict__ W, const float* __restrict__ R,
                        __bf16* __restrict__ dst, int N, int K) {
  int i = blockIdx.x * blockDim.x + threadIdx.x;
  if (i < N * K) {
    int n = i / K, k = i - n * K;
    size_t o = (size_t)n * (2 * K);
    dst[o + k]     = (__bf16)W[i];
    dst[o + K + k] = (__bf16)R[i];
  }
}

__global__ void bias_combine(const float* __restrict__ bi, const float* __restrict__ bh,
                             float* __restrict__ o, int n) {
  int i = blockIdx.x * blockDim.x + threadIdx.x;
  if (i < n) o[i] = bi[i] + bh[i];
}

__global__ void sigmoid_vec(const float* __restrict__ f, float* __restrict__ o, int n) {
  int i = blockIdx.x * blockDim.x + threadIdx.x;
  if (i < n) o[i] = 1.0f / (1.0f + __expf(-f[i]));
}

// ============================================================================
// Host launcher
// Inputs: 0:x 1:W0 2:R0 3:bi0 4:bh0 5:f0 6:W1 7:R1 8:bi1 9:bh1 10:f1
// Output: [out (T,B,H) | h_final (L,B,H) | c_final (L,B,H)] fp32
// ============================================================================
extern "C" void kernel_launch(void* const* d_in, const int* in_sizes, int n_in,
                              void* d_out, int out_size, void* d_ws, size_t ws_size,
                              hipStream_t stream) {
  (void)in_sizes; (void)n_in; (void)out_size; (void)ws_size;

  const float* x   = (const float*)d_in[0];
  const float* W0  = (const float*)d_in[1];
  const float* R0  = (const float*)d_in[2];
  const float* bi0 = (const float*)d_in[3];
  const float* bh0 = (const float*)d_in[4];
  const float* f0  = (const float*)d_in[5];
  const float* W1  = (const float*)d_in[6];
  const float* R1  = (const float*)d_in[7];
  const float* bi1 = (const float*)d_in[8];
  const float* bh1 = (const float*)d_in[9];
  const float* f1  = (const float*)d_in[10];

  // Workspace carve-up (256B aligned chunks)
  size_t cur = 0;
  auto take = [&](size_t bytes) -> char* {
    cur = (cur + 255) & ~(size_t)255;
    char* p = (char*)d_ws + cur;
    cur += bytes;
    return p;
  };
  __bf16* xb  = (__bf16*)take((size_t)TT * BB * II * 2);   // 64 MB
  __bf16* rw0 = (__bf16*)take((size_t)GH * K2 * 2);        // 12 MB
  __bf16* rw1 = (__bf16*)take((size_t)GH * K2 * 2);        // 12 MB
  __bf16* h0a = (__bf16*)take((size_t)BB * HH * 2);
  __bf16* h0b = (__bf16*)take((size_t)BB * HH * 2);
  __bf16* h1a = (__bf16*)take((size_t)BB * HH * 2);
  __bf16* h1b = (__bf16*)take((size_t)BB * HH * 2);
  float* c0    = (float*)take((size_t)BB * HH * 4);
  float* c1    = (float*)take((size_t)BB * HH * 4);
  float* bias0 = (float*)take((size_t)GH * 4);
  float* bias1 = (float*)take((size_t)GH * 4);
  float* fg0   = (float*)take((size_t)HH * 4);
  float* fg1   = (float*)take((size_t)HH * 4);
  unsigned* cnt = (unsigned*)take(256);

  // Zero recurrent state + barrier counter (graph-capturable async memsets)
  hipMemsetAsync(h0a, 0, (size_t)BB * HH * 2, stream);
  hipMemsetAsync(h0b, 0, (size_t)BB * HH * 2, stream);
  hipMemsetAsync(h1a, 0, (size_t)BB * HH * 2, stream);
  hipMemsetAsync(h1b, 0, (size_t)BB * HH * 2, stream);
  hipMemsetAsync(c0, 0, (size_t)BB * HH * 4, stream);
  hipMemsetAsync(c1, 0, (size_t)BB * HH * 4, stream);
  hipMemsetAsync(cnt, 0, 256, stream);

  // One-time conversions / packing
  {
    int n = TT * BB * II;
    f32_to_bf16<<<(n + 255) / 256, 256, 0, stream>>>(x, xb, n);
  }
  {
    int n = GH * HH;
    pack_rw<<<(n + 255) / 256, 256, 0, stream>>>(W0, R0, rw0, GH, HH);
    pack_rw<<<(n + 255) / 256, 256, 0, stream>>>(W1, R1, rw1, GH, HH);
  }
  bias_combine<<<(GH + 255) / 256, 256, 0, stream>>>(bi0, bh0, bias0, GH);
  bias_combine<<<(GH + 255) / 256, 256, 0, stream>>>(bi1, bh1, bias1, GH);
  sigmoid_vec<<<(HH + 255) / 256, 256, 0, stream>>>(f0, fg0, HH);
  sigmoid_vec<<<(HH + 255) / 256, 256, 0, stream>>>(f1, fg1, HH);

  // Single persistent launch: the whole recurrence, zero further launch cost.
  sublstm_persistent<<<NBLK, 512, 0, stream>>>(
      xb, rw0, rw1, bias0, bias1, fg0, fg1, c0, c1,
      h0a, h0b, h1a, h1b, (float*)d_out, cnt);
}
